// GCNmasker_7043746365840
// MI455X (gfx1250) — matrix operand order repaired and checked by reading, not compiled, not run on hardware
//
#include <hip/hip_runtime.h>
#include <hip/hip_bf16.h>

// ---------------------------------------------------------------------------
// GCN masker pipeline for MI455X (gfx1250, wave32, WMMA).
//   h  = relu(BN(x) @ Wf + bf)                  -> WMMA f16 GEMM, K=256
//   3x: h = BN(h); hW = h @ Ws[i] (WMMA K=128);
//       h = dinv^2*hW + b  +  scatter(dinv_s*dinv_d*hW[src] -> dst); relu(<L-1)
//   score = sigmoid(dot(h[src], h[dst]))        -> wave32 shfl reduction
// GEMM: A tile (16 x K f16, 8KB) staged in LDS once per block, 8 waves share it.
// Scatter: one wave per edge, float4 per lane, 4x global_atomic_add_f32.
// ---------------------------------------------------------------------------

typedef __attribute__((ext_vector_type(16))) _Float16 v16h;
typedef __attribute__((ext_vector_type(8)))  _Float16 v8h;
typedef __attribute__((ext_vector_type(4)))  _Float16 v4h;
typedef __attribute__((ext_vector_type(8)))  float    v8f;

// ---- small utility kernels -------------------------------------------------

__global__ void zero2_kernel(float* a, float* b, int f) {
    int i = blockIdx.x * blockDim.x + threadIdx.x;
    if (i < f) { a[i] = 0.f; b[i] = 0.f; }
}

// per-column sum / sum-of-squares over rows; blockDim.x == F (128 or 256)
__global__ void colstats_kernel(const float* __restrict__ x, int n, int F,
                                float* __restrict__ sum, float* __restrict__ sumsq) {
    int j = threadIdx.x;
    float s = 0.f, ss = 0.f;
    for (int r = blockIdx.x; r < n; r += gridDim.x) {
        float v = x[(size_t)r * F + j];
        s += v; ss += v * v;
    }
    atomicAdd(&sum[j], s);
    atomicAdd(&sumsq[j], ss);
}

// fold BN (training-mode, biased var) + gamma/beta into per-column scale/shift
__global__ void bn_coeff_kernel(const float* __restrict__ sum, const float* __restrict__ sumsq,
                                const float* __restrict__ g, const float* __restrict__ b,
                                int n, int F, float* __restrict__ scale, float* __restrict__ shift) {
    int j = blockIdx.x * blockDim.x + threadIdx.x;
    if (j >= F) return;
    float inv_n = 1.f / (float)n;
    float mu  = sum[j] * inv_n;
    float var = sumsq[j] * inv_n - mu * mu;
    float a   = g[j] * rsqrtf(var + 1e-5f);
    scale[j] = a;
    shift[j] = b[j] - mu * a;
}

// y_f16[4i..4i+3] = (x*scale + shift), BN fused into the f32->f16 downconvert
__global__ void bn_apply_f16_kernel(const float4* __restrict__ x4,
                                    const float* __restrict__ scale,
                                    const float* __restrict__ shift,
                                    v4h* __restrict__ y4, size_t total4, int F4) {
    for (size_t i = (size_t)blockIdx.x * blockDim.x + threadIdx.x; i < total4;
         i += (size_t)gridDim.x * blockDim.x) {
        int j = (int)(i % (size_t)F4) * 4;
        float4 xv = x4[i];
        v4h o;
        o[0] = (_Float16)(xv.x * scale[j + 0] + shift[j + 0]);
        o[1] = (_Float16)(xv.y * scale[j + 1] + shift[j + 1]);
        o[2] = (_Float16)(xv.z * scale[j + 2] + shift[j + 2]);
        o[3] = (_Float16)(xv.w * scale[j + 3] + shift[j + 3]);
        y4[i] = o;
    }
}

// W[K x H] f32 row-major -> Bt[H x K] f16 (column-major weights for WMMA B frags)
__global__ void w_f16t_kernel(const float* __restrict__ W, _Float16* __restrict__ Bt,
                              int K, int H) {
    int idx = blockIdx.x * blockDim.x + threadIdx.x;
    if (idx >= K * H) return;
    int k = idx / H, n = idx % H;
    Bt[(size_t)n * K + k] = (_Float16)W[idx];
}

// ---- WMMA GEMM: C[N x H] = A_f16[N x K] @ W_f16t[H x K]^T (+bias)(+relu) ----
// block = 256 threads = 8 waves; wave w computes the 16x16 tile (blockIdx.x, w).
// A tile (16 x K) staged once in LDS; fragments read via ds_load_b128.
template <int K>
__global__ __launch_bounds__(256) void wmma_gemm_kernel(
    const _Float16* __restrict__ A, const _Float16* __restrict__ Bt,
    const float* __restrict__ bias, float* __restrict__ C,
    int nrows, int H, int flags /* bit0: +bias, bit1: relu */) {
    __shared__ _Float16 As[16 * K];                  // 8KB (K=256) / 4KB (K=128)
    const int tid   = threadIdx.x;
    const int lane  = tid & 31;
    const int wave  = tid >> 5;
    const int tileM = blockIdx.x;
    const int kg    = lane >> 4;                     // lane group (0/1)
    const int ncol  = wave * 16 + (lane & 15);       // C/B column this lane holds
    if ((tileM + 1) * 16 > nrows) return;            // uniform per block

    // cooperative stage of the 16 x K A tile (rows are contiguous in memory)
    {
        const _Float16* Ablk = A + (size_t)tileM * 16 * K;
        constexpr int CHUNKS = (16 * K) / 8;         // v8h chunks
        for (int c = tid; c < CHUNKS; c += 256)
            *(v8h*)(As + c * 8) = *(const v8h*)(Ablk + c * 8);
    }
    __syncthreads();

    const _Float16* Arow = As + (lane & 15) * K;
    v8f acc = {};
#pragma unroll
    for (int kb = 0; kb < K; kb += 32) {
        // A 16x32 f16 frag (ISA layout): lanes 0-15 K=[kb..+7],[kb+16..+23];
        // lanes 16-31 shifted by +8
        union { v16h v; v8h h[2]; } a;
        a.h[0] = *(const v8h*)(Arow + kb + kg * 8);
        a.h[1] = *(const v8h*)(Arow + kb + 16 + kg * 8);
        // B 32x16 f16 frag: same K layout as A (lanes 0-15 K=[kb..+7],[kb+16..+23]; hi lanes +8)
        union { v16h v; v8h h[2]; } b;
        b.h[0] = *(const v8h*)(Bt + (size_t)ncol * K + kb + kg * 8);
        b.h[1] = *(const v8h*)(Bt + (size_t)ncol * K + kb + 16 + kg * 8);
        v16h bfr = b.v;
        acc = __builtin_amdgcn_wmma_f32_16x16x32_f16(
            /*neg_a=*/false, a.v, /*neg_b=*/false, bfr,
            /*c_mod=*/(short)0, acc, /*reuse_a=*/false, /*reuse_b=*/false);
    }

    float bb = (flags & 1) ? bias[ncol] : 0.f;
#pragma unroll
    for (int v = 0; v < 8; ++v) {                    // C layout: VGPR v -> M = v (+8 hi lanes)
        int m = tileM * 16 + kg * 8 + v;
        float val = acc[v] + bb;
        if (flags & 2) val = fmaxf(val, 0.f);
        C[(size_t)m * H + ncol] = val;
    }
}

// ---- degree / symmetric GCN normalization ---------------------------------

__global__ void deg_init_kernel(float* dinv, int n) {
    int i = blockIdx.x * blockDim.x + threadIdx.x;
    if (i < n) dinv[i] = 1.f;                        // self loop
}
__global__ void deg_count_kernel(const long long* __restrict__ dst, int e, float* dinv) {
    int i = blockIdx.x * blockDim.x + threadIdx.x;
    if (i < e) atomicAdd(&dinv[dst[i]], 1.f);
}
__global__ void deg_fin_kernel(float* dinv, int n) {
    int i = blockIdx.x * blockDim.x + threadIdx.x;
    if (i < n) dinv[i] = rsqrtf(fmaxf(dinv[i], 1.f));
}

// out[v,:] = dinv[v]^2 * hW[v,:] + bias[:]   (self-loop term + conv bias), float4
__global__ void prop_init_kernel(const float4* __restrict__ hW4, const float* __restrict__ dinv,
                                 const float4* __restrict__ bias4, float4* __restrict__ out4,
                                 int n, int H4) {
    size_t total = (size_t)n * H4;
    for (size_t i = (size_t)blockIdx.x * blockDim.x + threadIdx.x; i < total;
         i += (size_t)gridDim.x * blockDim.x) {
        int v = (int)(i / (size_t)H4), j = (int)(i % (size_t)H4);
        float di = dinv[v];
        float w = di * di;
        float4 hv = hW4[i], bv = bias4[j];
        out4[i] = make_float4(w * hv.x + bv.x, w * hv.y + bv.y,
                              w * hv.z + bv.z, w * hv.w + bv.w);
    }
}

// one wave32 per edge; lane moves float4 (H=128 => 32 lanes x 4 feats)
__global__ __launch_bounds__(256) void prop_edge_kernel(
    const long long* __restrict__ src, const long long* __restrict__ dst,
    const float* __restrict__ dinv, const float* __restrict__ hW,
    float* __restrict__ out, int E, int H) {
    int lane = threadIdx.x & 31;
    int wid  = blockIdx.x * (blockDim.x >> 5) + (threadIdx.x >> 5);
    int nw   = gridDim.x * (blockDim.x >> 5);
    for (int e = wid; e < E; e += nw) {
        long long s = src[e], d = dst[e];          // uniform address -> broadcast
        float w = dinv[s] * dinv[d];
        float4 v = *((const float4*)(hW + (size_t)s * H) + lane);
        float* po = out + (size_t)d * H + lane * 4;
        atomicAdd(po + 0, w * v.x);
        atomicAdd(po + 1, w * v.y);
        atomicAdd(po + 2, w * v.z);
        atomicAdd(po + 3, w * v.w);
    }
}

__global__ void relu4_kernel(float4* x, size_t total4) {
    for (size_t i = (size_t)blockIdx.x * blockDim.x + threadIdx.x; i < total4;
         i += (size_t)gridDim.x * blockDim.x) {
        float4 v = x[i];
        x[i] = make_float4(fmaxf(v.x, 0.f), fmaxf(v.y, 0.f),
                           fmaxf(v.z, 0.f), fmaxf(v.w, 0.f));
    }
}

// one wave per edge: dot(h[src], h[dst]) over H=128 via float4 + shfl_xor reduce
__global__ __launch_bounds__(256) void link_kernel(const long long* __restrict__ src,
                                                   const long long* __restrict__ dst,
                                                   const float* __restrict__ h,
                                                   float* __restrict__ out, int E, int H) {
    int lane = threadIdx.x & 31;
    int wid  = blockIdx.x * (blockDim.x >> 5) + (threadIdx.x >> 5);
    int nw   = gridDim.x * (blockDim.x >> 5);
    for (int e = wid; e < E; e += nw) {
        float4 a = *((const float4*)(h + (size_t)src[e] * H) + lane);
        float4 b = *((const float4*)(h + (size_t)dst[e] * H) + lane);
        float acc = a.x * b.x + a.y * b.y + a.z * b.z + a.w * b.w;
        for (int m = 16; m; m >>= 1) acc += __shfl_xor(acc, m, 32);
        if (lane == 0) out[e] = 1.f / (1.f + __expf(-acc));
    }
}

// ---------------------------------------------------------------------------

extern "C" void kernel_launch(void* const* d_in, const int* in_sizes, int n_in,
                              void* d_out, int out_size, void* d_ws, size_t ws_size,
                              hipStream_t stream) {
    const float*     x      = (const float*)d_in[0];
    const long long* ei     = (const long long*)d_in[1];   // int64 [2, E]
    const float*     bnfg   = (const float*)d_in[2];
    const float*     bnfb   = (const float*)d_in[3];
    const float*     Wf     = (const float*)d_in[4];
    const float*     bf     = (const float*)d_in[5];
    const float*     bnsg   = (const float*)d_in[6];
    const float*     bnsb   = (const float*)d_in[7];
    const float*     Ws     = (const float*)d_in[8];
    const float*     bs     = (const float*)d_in[9];

    const int FIN = 256, H = 128, L = 3;
    const int N = in_sizes[0] / FIN;
    const int E = in_sizes[1] / 2;
    const long long* srcI = ei;
    const long long* dstI = ei + E;

    // carve workspace (256B aligned)
    char* wsb = (char*)d_ws;
    size_t off = 0;
    auto carve = [&](size_t bytes) -> char* {
        char* p = wsb + off;
        off = (off + bytes + 255) & ~(size_t)255;
        return p;
    };
    float*    sum   = (float*)carve(FIN * 4);
    float*    sumsq = (float*)carve(FIN * 4);
    float*    scale = (float*)carve(FIN * 4);
    float*    shift = (float*)carve(FIN * 4);
    float*    dinv  = (float*)carve((size_t)N * 4);
    _Float16* A16   = (_Float16*)carve((size_t)N * FIN * 2);
    _Float16* W16   = (_Float16*)carve((size_t)FIN * H * 2);
    float*    hcur  = (float*)carve((size_t)N * H * 4);
    float*    htmp  = (float*)carve((size_t)N * H * 4);
    (void)ws_size; (void)n_in; (void)out_size;

    const int TB = 256;
    const size_t NH  = (size_t)N * H;
    const size_t NH4 = NH / 4;
    const int gNH4 = (int)((NH4 + TB - 1) / TB);
    const int gEwave = (E + (TB >> 5) - 1) / (TB >> 5);    // one wave per edge

    // ---- feature BN + conv_feat GEMM (+bias, +relu) ----
    zero2_kernel<<<1, FIN, 0, stream>>>(sum, sumsq, FIN);
    colstats_kernel<<<512, FIN, 0, stream>>>(x, N, FIN, sum, sumsq);
    bn_coeff_kernel<<<1, FIN, 0, stream>>>(sum, sumsq, bnfg, bnfb, N, FIN, scale, shift);
    bn_apply_f16_kernel<<<2048, TB, 0, stream>>>((const float4*)x, scale, shift,
                                                 (v4h*)A16, (size_t)N * FIN / 4, FIN / 4);
    w_f16t_kernel<<<(FIN * H + TB - 1) / TB, TB, 0, stream>>>(Wf, W16, FIN, H);
    wmma_gemm_kernel<256><<<(N + 15) / 16, TB, 0, stream>>>(A16, W16, bf, hcur, N, H, 3);

    // ---- degree -> dinv (self loop included) ----
    deg_init_kernel<<<(N + TB - 1) / TB, TB, 0, stream>>>(dinv, N);
    deg_count_kernel<<<(E + TB - 1) / TB, TB, 0, stream>>>(dstI, E, dinv);
    deg_fin_kernel<<<(N + TB - 1) / TB, TB, 0, stream>>>(dinv, N);

    // ---- 3 GCN conv layers ----
    for (int i = 0; i < L; ++i) {
        zero2_kernel<<<1, H, 0, stream>>>(sum, sumsq, H);
        colstats_kernel<<<512, H, 0, stream>>>(hcur, N, H, sum, sumsq);
        bn_coeff_kernel<<<1, H, 0, stream>>>(sum, sumsq, bnsg + i * H, bnsb + i * H, N, H, scale, shift);
        bn_apply_f16_kernel<<<2048, TB, 0, stream>>>((const float4*)hcur, scale, shift,
                                                     (v4h*)A16, NH4, H / 4);
        w_f16t_kernel<<<(H * H + TB - 1) / TB, TB, 0, stream>>>(Ws + (size_t)i * H * H, W16, H, H);
        wmma_gemm_kernel<128><<<(N + 15) / 16, TB, 0, stream>>>(A16, W16, nullptr, htmp, N, H, 0);
        // propagation: init with self-loop + bias, then per-edge wave scatter-add
        prop_init_kernel<<<gNH4, TB, 0, stream>>>((const float4*)htmp, dinv,
                                                  (const float4*)(bs + i * H),
                                                  (float4*)hcur, N, H / 4);
        prop_edge_kernel<<<gEwave, TB, 0, stream>>>(srcI, dstI, dinv, htmp, hcur, E, H);
        if (i < L - 1) relu4_kernel<<<gNH4, TB, 0, stream>>>((float4*)hcur, NH4);
    }

    // ---- link scoring ----
    link_kernel<<<4096, TB, 0, stream>>>(srcI, dstI, hcur, (float*)d_out, E, H);
}
